// MultiModalLSTM_87411174408858
// MI455X (gfx1250) — compile-verified, matrix operand order checked by reading, not run on hardware
//
#include <hip/hip_runtime.h>
#include <hip/hip_bf16.h>
#include <math.h>

// ---------------------------------------------------------------------------
// MultiModalLSTM on MI455X (gfx1250): batch-parallel recurrence with bf16 WMMA.
//   B=4096, T=50, IN=4, H=256, K=3 modes, FS=6.
// Each workgroup owns 16 batch rows and runs the full encoder+decoder time
// loop with state in LDS; recurrent matmuls are 16x1024x256 GEMMs done with
// v_wmma_f32_16x16x32_bf16 against pre-packed bf16 weights (L2-resident).
// This revision: fast activations (v_rcp_f32 / exp-based tanh) and pointer-
// bump B/A addressing in the WMMA loop.
// ---------------------------------------------------------------------------

#define BATCH   4096
#define TSTEPS  50
#define INDIM   4
#define HDIM    256
#define G4H     1024      // 4*H
#define KMODES  3
#define FSTEPS  6
#define MTILE   16        // batch rows per workgroup
#define KT      (HDIM/32) // 8 k-tiles of 32

typedef __attribute__((ext_vector_type(16))) __bf16 v16bf;
typedef __attribute__((ext_vector_type(8)))  float  v8f;

static __device__ __forceinline__ unsigned short f2bf(float f) {
    unsigned int u = __float_as_uint(f);
    unsigned int r = (u + 0x7FFFu + ((u >> 16) & 1u)) >> 16;
    return (unsigned short)r;
}
static __device__ __forceinline__ float bf2f(unsigned short h) {
    return __uint_as_float(((unsigned int)h) << 16);
}
// fast activations: v_rcp_f32 + v_exp_f32 only (no IEEE div sequences)
static __device__ __forceinline__ float sigm(float v) {
    return __builtin_amdgcn_rcpf(1.0f + __expf(-v));
}
static __device__ __forceinline__ float tanh_fast(float v) {
    float e = __expf(2.0f * v);
    return (e - 1.0f) * __builtin_amdgcn_rcpf(e + 1.0f);
}

// ---------------------------------------------------------------------------
// Pack a row-major fp32 weight W[N][Kdim] (gate = W @ h, i.e. B[k][n]=W[n][k])
// into bf16 WMMA B-fragments: dst[((nt*KT + kt)*32 + lane)*16 + j]
//   n = nt*16 + (lane&15), k = kt*32 + (lane>>4)*16 + j
// ---------------------------------------------------------------------------
__global__ void pack_bf16_kernel(const float* __restrict__ W,
                                 unsigned short* __restrict__ dst,
                                 int N, int Kdim) {
    int idx = blockIdx.x * blockDim.x + threadIdx.x;
    int total = N * Kdim;
    if (idx >= total) return;
    int j    = idx & 15;
    int l    = (idx >> 4) & 31;
    int tile = idx >> 9;
    int kt   = tile % (Kdim / 32);
    int nt   = tile / (Kdim / 32);
    int n = nt * 16 + (l & 15);
    int k = kt * 32 + ((l >> 4) << 4) + j;
    dst[idx] = f2bf(W[(size_t)n * Kdim + k]);
}

// ---------------------------------------------------------------------------
// Wave GEMM with register-resident init:
//   g = (bih + bhh) + Wx * xin  +  A0 x B0  [+ A1 x B1]
// 2 groups of 4 16-wide output tiles per wave (low VGPR pressure, no spills);
// kt loop not unrolled; B/A addressed by pointer bump + immediate offsets.
// ---------------------------------------------------------------------------
template<int IDIM>
static __device__ __forceinline__ void gate_gemm(
        float (*gbuf)[G4H],
        const unsigned short (*A0)[HDIM], const unsigned short* __restrict__ B0,
        const unsigned short (*A1)[HDIM], const unsigned short* __restrict__ B1,
        const float* __restrict__ bih, const float* __restrict__ bhh,
        const float* __restrict__ Wx,   // [G4H][IDIM] row-major (nullptr if IDIM==0)
        const float* xin, int xstride,  // input row m: xin[m*xstride + i]
        int wave, int lane) {
    const int row0 = 8 * (lane >> 4);
    const int colL = lane & 15;
    const int kofs = (lane >> 4) * 16;

    // per-lane copy of the 8 input rows this half-wave touches
    float xr[8][IDIM > 0 ? IDIM : 1];
    if (IDIM > 0) {
#pragma unroll
        for (int r = 0; r < 8; ++r)
#pragma unroll
            for (int i = 0; i < IDIM; ++i)
                xr[r][i] = xin[(row0 + r) * xstride + i];
    }

#pragma unroll 1
    for (int grp = 0; grp < 2; ++grp) {
        const int ntbase = wave * 8 + grp * 4;
        v8f c[4];
        // init: bias (column only) + input outer-product term (row x column)
#pragma unroll
        for (int nt = 0; nt < 4; ++nt) {
            int col = (ntbase + nt) * 16 + colL;
            float base = bih[col] + bhh[col];
            if (IDIM > 0) {
                float w[IDIM > 0 ? IDIM : 1];
#pragma unroll
                for (int i = 0; i < IDIM; ++i) w[i] = Wx[col * IDIM + i];
#pragma unroll
                for (int r = 0; r < 8; ++r) {
                    float acc = base;
#pragma unroll
                    for (int i = 0; i < IDIM; ++i) acc += w[i] * xr[r][i];
                    c[nt][r] = acc;
                }
            } else {
#pragma unroll
                for (int r = 0; r < 8; ++r) c[nt][r] = base;
            }
        }

        // ---- A0 x B0 ----
        {
            const unsigned short* bp = B0 + ((size_t)ntbase * KT * 32 + (size_t)lane) * 16;
            const unsigned short* ap = &A0[colL][kofs];
#pragma unroll 1
            for (int kt = 0; kt < KT; ++kt) {
                v16bf a = *(const v16bf*)ap;
                v16bf b[4];
#pragma unroll
                for (int nt = 0; nt < 4; ++nt)
                    b[nt] = *(const v16bf*)(bp + nt * (KT * 512));  // nt*8KB imm offset
#pragma unroll
                for (int nt = 0; nt < 4; ++nt)
                    c[nt] = __builtin_amdgcn_wmma_f32_16x16x32_bf16(
                                false, a, false, b[nt], (short)0, c[nt], false, false);
                bp += 512;   // next kt tile (1KB)
                ap += 32;    // next 32 K values in the LDS row
            }
        }
        // ---- optional A1 x B1 ----
        if (A1) {
            const unsigned short* bp = B1 + ((size_t)ntbase * KT * 32 + (size_t)lane) * 16;
            const unsigned short* ap = &A1[colL][kofs];
#pragma unroll 1
            for (int kt = 0; kt < KT; ++kt) {
                v16bf a = *(const v16bf*)ap;
                v16bf b[4];
#pragma unroll
                for (int nt = 0; nt < 4; ++nt)
                    b[nt] = *(const v16bf*)(bp + nt * (KT * 512));
#pragma unroll
                for (int nt = 0; nt < 4; ++nt)
                    c[nt] = __builtin_amdgcn_wmma_f32_16x16x32_bf16(
                                false, a, false, b[nt], (short)0, c[nt], false, false);
                bp += 512;
                ap += 32;
            }
        }
        // store gate fragments for the cross-wave pointwise phase
#pragma unroll
        for (int nt = 0; nt < 4; ++nt) {
            int col = (ntbase + nt) * 16 + colL;
#pragma unroll
            for (int r = 0; r < 8; ++r) gbuf[row0 + r][col] = c[nt][r];
        }
    }
}

static __device__ __forceinline__ void lstm_pointwise(
        float (*gbuf)[G4H], float (*cbuf)[HDIM],
        unsigned short (*hbuf)[HDIM], int tid) {
    for (int e = tid; e < MTILE * HDIM; e += 256) {
        int m = e >> 8, n = e & (HDIM - 1);
        float gi = gbuf[m][n];
        float gf = gbuf[m][n + HDIM];
        float gg = gbuf[m][n + 2 * HDIM];
        float go = gbuf[m][n + 3 * HDIM];
        float cc = sigm(gf) * cbuf[m][n] + sigm(gi) * tanh_fast(gg);
        cbuf[m][n] = cc;
        hbuf[m][n] = f2bf(sigm(go) * tanh_fast(cc));
    }
}

// packed-weight slot ids in workspace
#define PK_ENC_WHH0 0
#define PK_ENC_WIH1 1
#define PK_ENC_WHH1 2
#define PK_DEC(k, which) (3 + (k) * 3 + (which))  // which: 0=Whh0 1=Wih1 2=Whh1
#define MAT_ELEMS ((size_t)G4H * HDIM)

__global__ __launch_bounds__(256, 1)
void MultiModalLSTM_87411174408858_kernel(
        const float* __restrict__ x,
        const float* __restrict__ enc_Wih0,
        const float* __restrict__ enc_bih0, const float* __restrict__ enc_bhh0,
        const float* __restrict__ enc_bih1, const float* __restrict__ enc_bhh1,
        const float* __restrict__ dec_Wih0,
        const float* __restrict__ dec_bih0, const float* __restrict__ dec_bhh0,
        const float* __restrict__ dec_bih1, const float* __restrict__ dec_bhh1,
        const float* __restrict__ head_W,  const float* __restrict__ head_b,
        const float* __restrict__ conf_W,  const float* __restrict__ conf_b,
        const unsigned short* __restrict__ wpk,
        float* __restrict__ out_modes, float* __restrict__ out_conf) {
    extern __shared__ unsigned char smem[];
    unsigned char* p = smem;
    float (*gbuf)[G4H]            = (float (*)[G4H])p;          p += sizeof(float) * MTILE * G4H;
    float (*c0)[HDIM]             = (float (*)[HDIM])p;         p += sizeof(float) * MTILE * HDIM;
    float (*c1)[HDIM]             = (float (*)[HDIM])p;         p += sizeof(float) * MTILE * HDIM;
    unsigned short (*h0)[HDIM]    = (unsigned short (*)[HDIM])p; p += sizeof(unsigned short) * MTILE * HDIM;
    unsigned short (*h1)[HDIM]    = (unsigned short (*)[HDIM])p; p += sizeof(unsigned short) * MTILE * HDIM;
    float (*c0s)[HDIM]            = (float (*)[HDIM])p;         p += sizeof(float) * MTILE * HDIM;
    float (*c1s)[HDIM]            = (float (*)[HDIM])p;         p += sizeof(float) * MTILE * HDIM;
    unsigned short (*h0s)[HDIM]   = (unsigned short (*)[HDIM])p; p += sizeof(unsigned short) * MTILE * HDIM;
    unsigned short (*h1s)[HDIM]   = (unsigned short (*)[HDIM])p; p += sizeof(unsigned short) * MTILE * HDIM;
    float (*inp)[2]               = (float (*)[2])p;            p += sizeof(float) * MTILE * 2;
    float (*logits)[4]            = (float (*)[4])p;            // MTILE*4 floats

    const int tid  = threadIdx.x;
    const int lane = tid & 31;
    const int wave = tid >> 5;
    const int wgbase = blockIdx.x * MTILE;   // first batch row of this tile

    // zero-init recurrent state
    for (int e = tid; e < MTILE * HDIM; e += 256) {
        int m = e >> 8, n = e & (HDIM - 1);
        c0[m][n] = 0.f; c1[m][n] = 0.f; h0[m][n] = 0; h1[m][n] = 0;
    }
    __syncthreads();

    // ------------------------------ encoder ------------------------------
    for (int t = 0; t < TSTEPS; ++t) {
        // layer0: g = b + Wih0*x_t + Whh0*h0   (input term folded into C init)
        gate_gemm<INDIM>(gbuf, h0, wpk + PK_ENC_WHH0 * MAT_ELEMS, nullptr, nullptr,
                         enc_bih0, enc_bhh0, enc_Wih0,
                         x + ((size_t)wgbase * TSTEPS + t) * INDIM, TSTEPS * INDIM,
                         wave, lane);
        __syncthreads();
        lstm_pointwise(gbuf, c0, h0, tid);
        __syncthreads();

        // layer1: g = b + Wih1*h0 + Whh1*h1
        gate_gemm<0>(gbuf, h0, wpk + PK_ENC_WIH1 * MAT_ELEMS,
                           h1, wpk + PK_ENC_WHH1 * MAT_ELEMS,
                     enc_bih1, enc_bhh1, nullptr, nullptr, 0, wave, lane);
        __syncthreads();
        lstm_pointwise(gbuf, c1, h1, tid);
        __syncthreads();
    }

    // snapshot encoder final state for the K decoder modes
    for (int e = tid; e < MTILE * HDIM; e += 256) {
        int m = e >> 8, n = e & (HDIM - 1);
        c0s[m][n] = c0[m][n]; c1s[m][n] = c1[m][n];
        h0s[m][n] = h0[m][n]; h1s[m][n] = h1[m][n];
    }
    __syncthreads();

    // confidence head: softmax(h1 @ conf_W.T + conf_b)
    if (tid < MTILE * KMODES) {
        int m = tid / KMODES, k = tid % KMODES;
        float acc = conf_b[k];
        for (int n = 0; n < HDIM; ++n) acc += bf2f(h1s[m][n]) * conf_W[k * HDIM + n];
        logits[m][k] = acc;
    }
    __syncthreads();
    if (tid < MTILE) {
        float a = logits[tid][0], b = logits[tid][1], c = logits[tid][2];
        float mx = fmaxf(a, fmaxf(b, c));
        float ea = __expf(a - mx), eb = __expf(b - mx), ec = __expf(c - mx);
        float rs = __builtin_amdgcn_rcpf(ea + eb + ec);
        float* dst = out_conf + (size_t)(wgbase + tid) * KMODES;
        dst[0] = ea * rs; dst[1] = eb * rs; dst[2] = ec * rs;
    }

    // ------------------------------ decoder ------------------------------
    for (int k = 0; k < KMODES; ++k) {
        // restore encoder state + initial input x[:, -1, :2]
        for (int e = tid; e < MTILE * HDIM; e += 256) {
            int m = e >> 8, n = e & (HDIM - 1);
            c0[m][n] = c0s[m][n]; c1[m][n] = c1s[m][n];
            h0[m][n] = h0s[m][n]; h1[m][n] = h1s[m][n];
        }
        if (tid < MTILE * 2) {
            int m = tid >> 1, j = tid & 1;
            inp[m][j] = x[((size_t)(wgbase + m) * TSTEPS + (TSTEPS - 1)) * INDIM + j];
        }
        __syncthreads();

        for (int t = 0; t < FSTEPS; ++t) {
            // layer0: g = b + dec_Wih0[k]*inp + dec_Whh0[k]*h0
            gate_gemm<2>(gbuf, h0, wpk + PK_DEC(k, 0) * MAT_ELEMS, nullptr, nullptr,
                         dec_bih0 + (size_t)k * G4H, dec_bhh0 + (size_t)k * G4H,
                         dec_Wih0 + (size_t)k * G4H * 2,
                         (const float*)inp, 2, wave, lane);
            __syncthreads();
            lstm_pointwise(gbuf, c0, h0, tid);
            __syncthreads();

            // layer1: g = b + dec_Wih1[k]*h0 + dec_Whh1[k]*h1
            gate_gemm<0>(gbuf, h0, wpk + PK_DEC(k, 1) * MAT_ELEMS,
                               h1, wpk + PK_DEC(k, 2) * MAT_ELEMS,
                         dec_bih1 + (size_t)k * G4H, dec_bhh1 + (size_t)k * G4H,
                         nullptr, nullptr, 0, wave, lane);
            __syncthreads();
            lstm_pointwise(gbuf, c1, h1, tid);
            __syncthreads();

            // prediction head: pred = h1 @ head_W[k].T + head_b[k]  (H -> 2)
            if (tid < MTILE * 2) {
                int m = tid >> 1, j = tid & 1;
                float acc = head_b[k * 2 + j];
                const float* wr = head_W + ((size_t)k * 2 + j) * HDIM;
                for (int n = 0; n < HDIM; ++n) acc += bf2f(h1[m][n]) * wr[n];
                inp[m][j] = acc;
                out_modes[(((size_t)(wgbase + m) * KMODES + k) * FSTEPS + t) * 2 + j] = acc;
            }
            __syncthreads();
        }
    }
}

extern "C" void kernel_launch(void* const* d_in, const int* in_sizes, int n_in,
                              void* d_out, int out_size, void* d_ws, size_t ws_size,
                              hipStream_t stream) {
    (void)in_sizes; (void)n_in; (void)out_size; (void)ws_size;
    const float* x        = (const float*)d_in[0];
    const float* enc_Wih0 = (const float*)d_in[1];
    const float* enc_Whh0 = (const float*)d_in[2];
    const float* enc_bih0 = (const float*)d_in[3];
    const float* enc_bhh0 = (const float*)d_in[4];
    const float* enc_Wih1 = (const float*)d_in[5];
    const float* enc_Whh1 = (const float*)d_in[6];
    const float* enc_bih1 = (const float*)d_in[7];
    const float* enc_bhh1 = (const float*)d_in[8];
    const float* dec_Wih0 = (const float*)d_in[9];
    const float* dec_Whh0 = (const float*)d_in[10];
    const float* dec_bih0 = (const float*)d_in[11];
    const float* dec_bhh0 = (const float*)d_in[12];
    const float* dec_Wih1 = (const float*)d_in[13];
    const float* dec_Whh1 = (const float*)d_in[14];
    const float* dec_bih1 = (const float*)d_in[15];
    const float* dec_bhh1 = (const float*)d_in[16];
    const float* head_W   = (const float*)d_in[17];
    const float* head_b   = (const float*)d_in[18];
    const float* conf_W   = (const float*)d_in[19];
    const float* conf_b   = (const float*)d_in[20];

    unsigned short* wpk = (unsigned short*)d_ws;   // 12 packed 1024x256 bf16 matrices (~6MB)

    const int packThreads = 256;
    const int packBlocks  = (int)((MAT_ELEMS + packThreads - 1) / packThreads);
    pack_bf16_kernel<<<packBlocks, packThreads, 0, stream>>>(enc_Whh0, wpk + PK_ENC_WHH0 * MAT_ELEMS, G4H, HDIM);
    pack_bf16_kernel<<<packBlocks, packThreads, 0, stream>>>(enc_Wih1, wpk + PK_ENC_WIH1 * MAT_ELEMS, G4H, HDIM);
    pack_bf16_kernel<<<packBlocks, packThreads, 0, stream>>>(enc_Whh1, wpk + PK_ENC_WHH1 * MAT_ELEMS, G4H, HDIM);
    for (int k = 0; k < KMODES; ++k) {
        pack_bf16_kernel<<<packBlocks, packThreads, 0, stream>>>(
            dec_Whh0 + (size_t)k * G4H * HDIM, wpk + PK_DEC(k, 0) * MAT_ELEMS, G4H, HDIM);
        pack_bf16_kernel<<<packBlocks, packThreads, 0, stream>>>(
            dec_Wih1 + (size_t)k * G4H * HDIM, wpk + PK_DEC(k, 1) * MAT_ELEMS, G4H, HDIM);
        pack_bf16_kernel<<<packBlocks, packThreads, 0, stream>>>(
            dec_Whh1 + (size_t)k * G4H * HDIM, wpk + PK_DEC(k, 2) * MAT_ELEMS, G4H, HDIM);
    }

    float* out_modes = (float*)d_out;
    float* out_conf  = out_modes + (size_t)BATCH * KMODES * FSTEPS * 2;

    const size_t smem =
        sizeof(float) * MTILE * G4H +                 // gbuf
        4 * sizeof(float) * MTILE * HDIM +            // c0,c1,c0s,c1s
        4 * sizeof(unsigned short) * MTILE * HDIM +   // h0,h1,h0s,h1s
        sizeof(float) * MTILE * 2 +                   // inp
        sizeof(float) * MTILE * 4;                    // logits

    dim3 grid(BATCH / MTILE);
    dim3 block(256);
    MultiModalLSTM_87411174408858_kernel<<<grid, block, smem, stream>>>(
        x, enc_Wih0, enc_bih0, enc_bhh0, enc_bih1, enc_bhh1,
        dec_Wih0, dec_bih0, dec_bhh0, dec_bih1, dec_bhh1,
        head_W, head_b, conf_W, conf_b, wpk, out_modes, out_conf);
}
